// Transformer_11690900979973
// MI455X (gfx1250) — compile-verified, matrix operand order checked by reading, not run on hardware
//
#include <hip/hip_runtime.h>

#define S_LEN  1024
#define BATCH  4
#define DMODEL 1024
#define HKDIM  1024
#define MFFN   4096
#define NLAYER 6
#define VOCAB  32000
#define TOK    (S_LEN * BATCH)   // 4096 tokens

typedef __attribute__((ext_vector_type(16))) _Float16 v16h;
typedef __attribute__((ext_vector_type(8)))  _Float16 v8h;
typedef __attribute__((ext_vector_type(8)))  float    v8f;
typedef long long ll;

// ---------------------------------------------------------------------------
// WMMA fragment gather, per ISA 7.12.2 (16-bit A/B): lane = {m|n = lane&15,
// khalf = lane>>4}; elements 0..7 -> k = kb+8*khalf+e, elements 8..15 ->
// k = kb+16+8*khalf+(e-8)  => two contiguous b128 loads at base+imm offsets.
// ---------------------------------------------------------------------------
__device__ __forceinline__ v16h frag(const _Float16* __restrict__ p, int kb) {
  v8h lo = *(const v8h*)(p + kb);
  v8h hi = *(const v8h*)(p + kb + 16);
  return __builtin_shufflevector(lo, hi, 0,1,2,3,4,5,6,7,8,9,10,11,12,13,14,15);
}

#define WMMA(a, b, c) \
  __builtin_amdgcn_wmma_f32_16x16x32_f16(false, a, false, b, (short)0, c, false, false)

// C/D layout: element r -> m = m0 + 8*(lane>>4) + r, n = n0 + (lane&15)
template <bool HB, bool HR, bool HC>
__device__ __forceinline__ void wmma_store(const v8f& c, float* __restrict__ C,
                                           _Float16* __restrict__ Ch,
                                           const float* __restrict__ res,
                                           const float* __restrict__ bias,
                                           int m0, int n0, int ldc, int lane) {
  int n  = n0 + (lane & 15);
  int mb = m0 + ((lane >> 4) << 3);
  float bv = HB ? bias[n] : 0.0f;
#pragma unroll
  for (int r = 0; r < 8; ++r) {
    size_t idx = (size_t)(mb + r) * ldc + n;
    float v = c[r] + bv;
    if (HR) v += res[idx];
    C[idx] = v;
    if (HC) Ch[idx] = (_Float16)v;
  }
}

// ---------------------------------------------------------------------------
// Batched GEMM: C = A[M,K] * Bt[N,K]^T (+bias[n]) (+res) (+f16 copy).
// 4 waves/block -> 64x128 block tile, 32x64 per wave: 8 v_wmma per K=32 step
// against 12 b128 loads (~21 flop/B from L2). Software-pipelined: next
// K-step's 6 fragments are loaded before the current step's WMMAs; unroll 2
// makes the double-buffer copies vanish. Epilogue is branch-free (templated).
// Requires M % 64 == 0, N % 128 == 0, K % 64 == 0 (all call sites comply).
// ---------------------------------------------------------------------------
template <bool HB, bool HR, bool HC>
__global__ __launch_bounds__(128) void k_gemm(
    const _Float16* __restrict__ A, const _Float16* __restrict__ Bt,
    const float* __restrict__ bias, const float* __restrict__ res,
    float* __restrict__ C, _Float16* __restrict__ Ch,
    int K, int lda, int ldb, int ldc, ll sA, ll sB, ll sC, ll sR) {
  int bz = blockIdx.z;
  A  += (size_t)bz * sA;
  Bt += (size_t)bz * sB;
  C  += (size_t)bz * sC;
  if (HC) Ch  += (size_t)bz * sC;
  if (HR) res += (size_t)bz * sR;

  int wave = threadIdx.x >> 5;
  int lane = threadIdx.x & 31;
  int row  = lane & 15;
  int khf  = lane >> 4;
  int m0 = blockIdx.y * 64  + (wave >> 1) * 32;
  int n0 = blockIdx.x * 128 + (wave & 1) * 64;

  // one base pointer per fragment stream; K advances via immediate offsets
  const _Float16* pa0 = A + (size_t)(m0 + row) * lda + khf * 8;
  const _Float16* pa1 = pa0 + (size_t)16 * lda;
  const _Float16* pb0 = Bt + (size_t)(n0 + row) * ldb + khf * 8;
  const _Float16* pb1 = pb0 + (size_t)16 * ldb;
  const _Float16* pb2 = pb0 + (size_t)32 * ldb;
  const _Float16* pb3 = pb0 + (size_t)48 * ldb;

  v8f acc[2][4];
#pragma unroll
  for (int mi = 0; mi < 2; ++mi)
#pragma unroll
    for (int j = 0; j < 4; ++j) acc[mi][j] = v8f{};

  v16h a0 = frag(pa0, 0), a1 = frag(pa1, 0);
  v16h b0 = frag(pb0, 0), b1 = frag(pb1, 0), b2 = frag(pb2, 0), b3 = frag(pb3, 0);

#pragma unroll 2
  for (int kb = 32; kb < K; kb += 32) {
    v16h na0 = frag(pa0, kb), na1 = frag(pa1, kb);
    v16h nb0 = frag(pb0, kb), nb1 = frag(pb1, kb);
    v16h nb2 = frag(pb2, kb), nb3 = frag(pb3, kb);
    acc[0][0] = WMMA(a0, b0, acc[0][0]);
    acc[1][0] = WMMA(a1, b0, acc[1][0]);
    acc[0][1] = WMMA(a0, b1, acc[0][1]);
    acc[1][1] = WMMA(a1, b1, acc[1][1]);
    acc[0][2] = WMMA(a0, b2, acc[0][2]);
    acc[1][2] = WMMA(a1, b2, acc[1][2]);
    acc[0][3] = WMMA(a0, b3, acc[0][3]);
    acc[1][3] = WMMA(a1, b3, acc[1][3]);
    a0 = na0; a1 = na1; b0 = nb0; b1 = nb1; b2 = nb2; b3 = nb3;
  }
  acc[0][0] = WMMA(a0, b0, acc[0][0]);
  acc[1][0] = WMMA(a1, b0, acc[1][0]);
  acc[0][1] = WMMA(a0, b1, acc[0][1]);
  acc[1][1] = WMMA(a1, b1, acc[1][1]);
  acc[0][2] = WMMA(a0, b2, acc[0][2]);
  acc[1][2] = WMMA(a1, b2, acc[1][2]);
  acc[0][3] = WMMA(a0, b3, acc[0][3]);
  acc[1][3] = WMMA(a1, b3, acc[1][3]);

#pragma unroll
  for (int mi = 0; mi < 2; ++mi)
#pragma unroll
    for (int j = 0; j < 4; ++j)
      wmma_store<HB, HR, HC>(acc[mi][j], C, Ch, res, bias,
                             m0 + 16 * mi, n0 + 16 * j, ldc, lane);
}

// ---------------------------------------------------------------------------
// Embedding: z[s*B+b, d] = relu(word_emb[x[s,b]] * sqrt(D) + pos_emb[s])
// ---------------------------------------------------------------------------
__global__ __launch_bounds__(256) void k_embed(const int* __restrict__ x,
                                               const float* __restrict__ we,
                                               const float* __restrict__ pe,
                                               float* __restrict__ z, float sqrtd) {
  int t = blockIdx.x;            // token in [0, S*B), layout [S,B]
  int s = t / BATCH;
  int tok = x[t];
  const float* w = we + (size_t)tok * DMODEL;
  const float* p = pe + (size_t)s * DMODEL;
  float* zr = z + (size_t)t * DMODEL;
  for (int i = threadIdx.x; i < DMODEL; i += 256)
    zr[i] = fmaxf(w[i] * sqrtd + p[i], 0.0f);
}

// swap token dims: in [d0,d1,D] -> out [d1,d0,D]; also emit f16 copy
__global__ __launch_bounds__(256) void k_permute(const float* __restrict__ in,
                                                 float* __restrict__ outf,
                                                 _Float16* __restrict__ outh,
                                                 int d0, int d1) {
  int t = blockIdx.x;                    // over d0*d1
  int i = t / d1, j = t % d1;
  const float* src = in + (size_t)t * DMODEL;
  size_t o = ((size_t)j * d0 + i) * DMODEL;
  for (int k = threadIdx.x; k < DMODEL; k += 256) {
    float v = src[k];
    outf[o + k] = v;
    outh[o + k] = (_Float16)v;
  }
}

__global__ void k_convert(const float* __restrict__ in, _Float16* __restrict__ out, ll n) {
  ll i = (ll)blockIdx.x * blockDim.x + threadIdx.x;
  ll stride = (ll)gridDim.x * blockDim.x;
  for (; i < n; i += stride) out[i] = (_Float16)in[i];
}

// f32 [R,C] -> f16 [C,R] tiled transpose (per batch)
__global__ __launch_bounds__(256) void k_transpose(const float* __restrict__ in,
                                                   _Float16* __restrict__ out,
                                                   int R, int Ccols, ll sIn, ll sOut) {
  __shared__ float tile[16][17];
  int b = blockIdx.z;
  in  += (size_t)b * sIn;
  out += (size_t)b * sOut;
  int c0 = blockIdx.x * 16, r0 = blockIdx.y * 16;
  int tx = threadIdx.x & 15, ty = threadIdx.x >> 4;
  tile[ty][tx] = in[(size_t)(r0 + ty) * Ccols + c0 + tx];
  __syncthreads();
  out[(size_t)(c0 + ty) * R + r0 + tx] = (_Float16)tile[tx][ty];
}

// row softmax (len n), scale applied pre-max; writes f16 probs for ctx GEMM
__global__ __launch_bounds__(256) void k_softmax(const float* __restrict__ s,
                                                 _Float16* __restrict__ p,
                                                 int n, float scale) {
  __shared__ float red[256];
  size_t row = blockIdx.x;
  const float* sr = s + row * n;
  _Float16* pr = p + row * n;
  int t = threadIdx.x;
  float mx = -3.4e38f;
  for (int i = t; i < n; i += 256) mx = fmaxf(mx, sr[i] * scale);
  red[t] = mx; __syncthreads();
  for (int o = 128; o; o >>= 1) { if (t < o) red[t] = fmaxf(red[t], red[t + o]); __syncthreads(); }
  mx = red[0]; __syncthreads();
  float sum = 0.0f;
  for (int i = t; i < n; i += 256) sum += __expf(sr[i] * scale - mx);
  red[t] = sum; __syncthreads();
  for (int o = 128; o; o >>= 1) { if (t < o) red[t] += red[t + o]; __syncthreads(); }
  float inv = 1.0f / red[0];
  for (int i = t; i < n; i += 256) pr[i] = (_Float16)(__expf(sr[i] * scale - mx) * inv);
}

// layernorm over D=1024; writes f32 + f16 outputs
__global__ __launch_bounds__(256) void k_layernorm(const float* __restrict__ a,
                                                   const float* __restrict__ g,
                                                   const float* __restrict__ beta,
                                                   float* __restrict__ of,
                                                   _Float16* __restrict__ oh) {
  __shared__ float red[256];
  size_t row = blockIdx.x;
  const float* ar = a + row * DMODEL;
  float* ofr = of + row * DMODEL;
  _Float16* ohr = oh + row * DMODEL;
  int t = threadIdx.x;
  float v[4];
  float s = 0.0f;
#pragma unroll
  for (int j = 0; j < 4; ++j) { v[j] = ar[t + 256 * j]; s += v[j]; }
  red[t] = s; __syncthreads();
  for (int o = 128; o; o >>= 1) { if (t < o) red[t] += red[t + o]; __syncthreads(); }
  float mu = red[0] * (1.0f / DMODEL);
  __syncthreads();
  float vs = 0.0f;
#pragma unroll
  for (int j = 0; j < 4; ++j) { float c = v[j] - mu; vs += c * c; }
  red[t] = vs; __syncthreads();
  for (int o = 128; o; o >>= 1) { if (t < o) red[t] += red[t + o]; __syncthreads(); }
  float inv = rsqrtf(red[0] * (1.0f / DMODEL) + 1e-5f);
#pragma unroll
  for (int j = 0; j < 4; ++j) {
    int idx = t + 256 * j;
    float o2 = (v[j] - mu) * inv * g[idx] + beta[idx];
    ofr[idx] = o2;
    ohr[idx] = (_Float16)o2;
  }
}

// odd layers: attention over 1024 groups of length-4 sequences (tiny -> VALU)
__global__ __launch_bounds__(256) void k_attn_small(const float* __restrict__ Q,
                                                    const float* __restrict__ Km,
                                                    const float* __restrict__ V,
                                                    float* __restrict__ ctx,
                                                    float scale) {
  __shared__ float red[256];
  __shared__ float sc[BATCH * BATCH];
  __shared__ float pw[BATCH * BATCH];
  int g = blockIdx.x, t = threadIdx.x;
  const float* Qg = Q  + (size_t)g * BATCH * HKDIM;
  const float* Kg = Km + (size_t)g * BATCH * HKDIM;
  const float* Vg = V  + (size_t)g * BATCH * HKDIM;
  float* Cg = ctx + (size_t)g * BATCH * HKDIM;
  for (int p = 0; p < BATCH * BATCH; ++p) {
    int i = p >> 2, j = p & 3;
    float s = 0.0f;
    for (int d = t; d < HKDIM; d += 256) s += Qg[i * HKDIM + d] * Kg[j * HKDIM + d];
    red[t] = s; __syncthreads();
    for (int o = 128; o; o >>= 1) { if (t < o) red[t] += red[t + o]; __syncthreads(); }
    if (t == 0) sc[p] = red[0] * scale;
    __syncthreads();
  }
  if (t < BATCH) {
    float mx = -3.4e38f;
    for (int j = 0; j < BATCH; ++j) mx = fmaxf(mx, sc[t * BATCH + j]);
    float sum = 0.0f;
    for (int j = 0; j < BATCH; ++j) sum += __expf(sc[t * BATCH + j] - mx);
    float invs = 1.0f / sum;
    for (int j = 0; j < BATCH; ++j) pw[t * BATCH + j] = __expf(sc[t * BATCH + j] - mx) * invs;
  }
  __syncthreads();
  for (int d = t; d < HKDIM; d += 256) {
#pragma unroll
    for (int i = 0; i < BATCH; ++i) {
      float acc = 0.0f;
#pragma unroll
      for (int j = 0; j < BATCH; ++j) acc += pw[i * BATCH + j] * Vg[j * HKDIM + d];
      Cg[i * HKDIM + d] = acc;
    }
  }
}

__global__ __launch_bounds__(256) void k_addvec(const float* __restrict__ a,
                                                const float* __restrict__ b,
                                                float* __restrict__ o, int n) {
  int i = blockIdx.x * 256 + threadIdx.x;
  if (i < n) o[i] = a[i] + b[i];
}

// in-place log_softmax over rows of length n (n=32000, L2-resident per row)
__global__ __launch_bounds__(256) void k_logsoftmax(float* __restrict__ lg, int n) {
  __shared__ float red[256];
  size_t row = blockIdx.x;
  float* r = lg + row * (size_t)n;
  int t = threadIdx.x;
  float mx = -3.4e38f;
  for (int i = t; i < n; i += 256) mx = fmaxf(mx, r[i]);
  red[t] = mx; __syncthreads();
  for (int o = 128; o; o >>= 1) { if (t < o) red[t] = fmaxf(red[t], red[t + o]); __syncthreads(); }
  mx = red[0]; __syncthreads();
  float s = 0.0f;
  for (int i = t; i < n; i += 256) s += __expf(r[i] - mx);
  red[t] = s; __syncthreads();
  for (int o = 128; o; o >>= 1) { if (t < o) red[t] += red[t + o]; __syncthreads(); }
  float lse = mx + __logf(red[0]);
  for (int i = t; i < n; i += 256) r[i] = r[i] - lse;
}

// host-side dispatch over the 8 epilogue variants (branch-free device code)
static void gemm(dim3 grid, hipStream_t st,
                 const _Float16* A, const _Float16* Bt,
                 const float* bias, const float* res,
                 float* C, _Float16* Ch,
                 int K, int lda, int ldb, int ldc,
                 ll sA = 0, ll sB = 0, ll sC = 0, ll sR = 0) {
  dim3 blk(128);
#define GL(HB, HR, HC) \
  k_gemm<HB, HR, HC><<<grid, blk, 0, st>>>(A, Bt, bias, res, C, Ch, K, lda, ldb, ldc, sA, sB, sC, sR)
  if (bias) {
    if (res) { if (Ch) GL(true, true, true);   else GL(true, true, false); }
    else     { if (Ch) GL(true, false, true);  else GL(true, false, false); }
  } else {
    if (res) { if (Ch) GL(false, true, true);  else GL(false, true, false); }
    else     { if (Ch) GL(false, false, true); else GL(false, false, false); }
  }
#undef GL
}

// ---------------------------------------------------------------------------
extern "C" void kernel_launch(void* const* d_in, const int* in_sizes, int n_in,
                              void* d_out, int out_size, void* d_ws, size_t ws_size,
                              hipStream_t stream) {
  (void)in_sizes; (void)n_in; (void)out_size; (void)ws_size;
  const int*   x     = (const int*)d_in[0];
  const float* wemb  = (const float*)d_in[1];
  const float* pemb  = (const float*)d_in[2];
  const float* wq    = (const float*)d_in[3];
  const float* wk    = (const float*)d_in[4];
  const float* wv    = (const float*)d_in[5];
  const float* wc    = (const float*)d_in[6];
  const float* ln1g  = (const float*)d_in[7];
  const float* ln1b  = (const float*)d_in[8];
  const float* w1    = (const float*)d_in[9];
  const float* b1    = (const float*)d_in[10];
  const float* w2    = (const float*)d_in[11];
  const float* b2    = (const float*)d_in[12];
  const float* ln2g  = (const float*)d_in[13];
  const float* ln2b  = (const float*)d_in[14];
  const float* decw  = (const float*)d_in[15];
  const float* decb  = (const float*)d_in[16];
  const float* obias = (const float*)d_in[17];
  float* out = (float*)d_out;

  char* ws = (char*)d_ws;
  size_t off = 0;
  auto take = [&](size_t bytes) -> char* {
    char* p = ws + off;
    off += (bytes + 255) & ~(size_t)255;
    return p;
  };
  float*    zbuf  = (float*)take((size_t)TOK * DMODEL * 4);
  float*    xbuf  = (float*)take((size_t)TOK * DMODEL * 4);
  _Float16* xh    = (_Float16*)take((size_t)TOK * DMODEL * 2);
  float*    qf    = (float*)take((size_t)TOK * HKDIM * 4);
  float*    kf    = (float*)take((size_t)TOK * HKDIM * 4);
  float*    vf    = (float*)take((size_t)TOK * HKDIM * 4);
  _Float16* qh16  = (_Float16*)take((size_t)TOK * HKDIM * 2);
  _Float16* kh16  = (_Float16*)take((size_t)TOK * HKDIM * 2);
  _Float16* vth   = (_Float16*)take((size_t)TOK * HKDIM * 2);   // [B][HK][S]
  float*    scb   = (float*)take((size_t)BATCH * S_LEN * S_LEN * 4);
  _Float16* ph    = (_Float16*)take((size_t)BATCH * S_LEN * S_LEN * 2);
  float*    ctxf  = (float*)take((size_t)TOK * HKDIM * 4);
  _Float16* ctxh  = (_Float16*)take((size_t)TOK * HKDIM * 2);
  float*    attf  = (float*)take((size_t)TOK * DMODEL * 4);
  float*    hf    = (float*)take((size_t)TOK * DMODEL * 4);
  _Float16* hh16  = (_Float16*)take((size_t)TOK * DMODEL * 2);
  float*    ff    = (float*)take((size_t)TOK * MFFN * 4);
  _Float16* fh16  = (_Float16*)take((size_t)TOK * MFFN * 2);
  _Float16* zh16  = (_Float16*)take((size_t)TOK * DMODEL * 2);
  _Float16* wqh   = (_Float16*)take((size_t)HKDIM * DMODEL * 2);
  _Float16* wkh   = (_Float16*)take((size_t)HKDIM * DMODEL * 2);
  _Float16* wvh   = (_Float16*)take((size_t)HKDIM * DMODEL * 2);
  _Float16* wch   = (_Float16*)take((size_t)DMODEL * HKDIM * 2);
  _Float16* w1h   = (_Float16*)take((size_t)MFFN * DMODEL * 2);
  _Float16* w2h   = (_Float16*)take((size_t)DMODEL * MFFN * 2);
  _Float16* dwh   = (_Float16*)take((size_t)VOCAB * DMODEL * 2);
  float*    biasc = (float*)take((size_t)VOCAB * 4);

  const float scale = 1.0f / 32.0f;        // 1/sqrt(D), faithful (not 1/sqrt(dk))

  k_embed<<<TOK, 256, 0, stream>>>(x, wemb, pemb, zbuf, 32.0f /*sqrt(1024)*/);

  for (int i = 0; i < NLAYER; ++i) {
    const bool even = (i % 2 == 0);
    int d0 = even ? S_LEN : BATCH;
    int d1 = even ? BATCH : S_LEN;
    // entry permute: z [d0,d1,D] -> x [d1,d0,D] (+f16)
    k_permute<<<TOK, 256, 0, stream>>>(zbuf, xbuf, xh, d0, d1);

    // per-layer weight downconverts (L2-resident afterwards)
    k_convert<<<2048, 256, 0, stream>>>(wq + (size_t)i * HKDIM * DMODEL, wqh, (ll)HKDIM * DMODEL);
    k_convert<<<2048, 256, 0, stream>>>(wk + (size_t)i * HKDIM * DMODEL, wkh, (ll)HKDIM * DMODEL);
    k_convert<<<2048, 256, 0, stream>>>(wv + (size_t)i * HKDIM * DMODEL, wvh, (ll)HKDIM * DMODEL);
    k_convert<<<2048, 256, 0, stream>>>(wc + (size_t)i * DMODEL * HKDIM, wch, (ll)DMODEL * HKDIM);
    k_convert<<<2048, 256, 0, stream>>>(w1 + (size_t)i * MFFN * DMODEL, w1h, (ll)MFFN * DMODEL);
    k_convert<<<2048, 256, 0, stream>>>(w2 + (size_t)i * DMODEL * MFFN, w2h, (ll)DMODEL * MFFN);

    // Q,K,V = x * W^T   [TOK, HK]; f16 copies fused for the WMMA attention path
    gemm(dim3(HKDIM / 128, TOK / 64, 1), stream, xh, wqh, nullptr, nullptr,
         qf, even ? qh16 : nullptr, DMODEL, DMODEL, DMODEL, HKDIM);
    gemm(dim3(HKDIM / 128, TOK / 64, 1), stream, xh, wkh, nullptr, nullptr,
         kf, even ? kh16 : nullptr, DMODEL, DMODEL, DMODEL, HKDIM);
    gemm(dim3(HKDIM / 128, TOK / 64, 1), stream, xh, wvh, nullptr, nullptr,
         vf, nullptr, DMODEL, DMODEL, DMODEL, HKDIM);

    if (even) {
      // 4 batches x seq 1024: WMMA attention
      k_transpose<<<dim3(HKDIM / 16, S_LEN / 16, BATCH), 256, 0, stream>>>(
          vf, vth, S_LEN, HKDIM, (ll)S_LEN * HKDIM, (ll)HKDIM * S_LEN);
      // scores = Q K^T / 32
      gemm(dim3(S_LEN / 128, S_LEN / 64, BATCH), stream, qh16, kh16, nullptr, nullptr,
           scb, nullptr, HKDIM, HKDIM, HKDIM, S_LEN,
           (ll)S_LEN * HKDIM, (ll)S_LEN * HKDIM, (ll)S_LEN * S_LEN, 0);
      k_softmax<<<BATCH * S_LEN, 256, 0, stream>>>(scb, ph, S_LEN, scale);
      // ctx = P V  (V^T provided as Bt); f16 copy fused for wc GEMM
      gemm(dim3(HKDIM / 128, S_LEN / 64, BATCH), stream, ph, vth, nullptr, nullptr,
           ctxf, ctxh, S_LEN, S_LEN, S_LEN, HKDIM,
           (ll)S_LEN * S_LEN, (ll)HKDIM * S_LEN, (ll)S_LEN * HKDIM, 0);
    } else {
      // 1024 groups x seq 4: VALU attention (negligible FLOPs)
      k_attn_small<<<S_LEN, 256, 0, stream>>>(qf, kf, vf, ctxf, scale);
      k_convert<<<2048, 256, 0, stream>>>(ctxf, ctxh, (ll)TOK * HKDIM);
    }

    // att = ctx * wc^T + x ; then ln1 -> h
    gemm(dim3(DMODEL / 128, TOK / 64, 1), stream, ctxh, wch, nullptr, xbuf,
         attf, nullptr, HKDIM, HKDIM, HKDIM, DMODEL);
    k_layernorm<<<TOK, 256, 0, stream>>>(attf, ln1g + (size_t)i * DMODEL,
                                         ln1b + (size_t)i * DMODEL, hf, hh16);

    // FFN (no nonlinearity, faithful): f = h w1^T + b1 ; f2 = f w2^T + b2 + h
    gemm(dim3(MFFN / 128, TOK / 64, 1), stream, hh16, w1h, b1 + (size_t)i * MFFN, nullptr,
         ff, fh16, DMODEL, DMODEL, DMODEL, MFFN);
    gemm(dim3(DMODEL / 128, TOK / 64, 1), stream, fh16, w2h, b2 + (size_t)i * DMODEL, hf,
         attf, nullptr, MFFN, MFFN, MFFN, DMODEL);
    // ln2 -> next z (f32) and f16 copy (decoder input after last layer)
    k_layernorm<<<TOK, 256, 0, stream>>>(attf, ln2g + (size_t)i * DMODEL,
                                         ln2b + (size_t)i * DMODEL, zbuf, zh16);
  }

  // decoder: logits = z dec_w^T + (dec_b + out_bias), then log_softmax
  k_convert<<<4096, 256, 0, stream>>>(decw, dwh, (ll)VOCAB * DMODEL);
  k_addvec<<<(VOCAB + 255) / 256, 256, 0, stream>>>(decb, obias, biasc, VOCAB);
  gemm(dim3(VOCAB / 128, TOK / 64, 1), stream, zh16, dwh, biasc, nullptr,
       out, nullptr, DMODEL, DMODEL, DMODEL, VOCAB);
  k_logsoftmax<<<TOK, 256, 0, stream>>>(out, VOCAB);
}
